// DecoderCell_11562051961566
// MI455X (gfx1250) — matrix operand order repaired and checked by reading, not compiled, not run on hardware
//
#include <hip/hip_runtime.h>
#include <hip/hip_bf16.h>
#include <math.h>

// ---------------------------------------------------------------------------
// Dims (fixed by reference): B=64, IN=512, H=512, A=512, L=2048, 2H=1024
// ---------------------------------------------------------------------------
#define BB   64
#define HH   512
#define AA   512
#define LL   2048
#define H2   1024

typedef __attribute__((ext_vector_type(16))) __bf16       v16bf;
typedef __attribute__((ext_vector_type(8)))  float        v8f;
typedef __attribute__((ext_vector_type(4)))  unsigned int u32x4;

__device__ __forceinline__ unsigned short f32_to_bf16(float f) {
  union { float f; unsigned u; } v; v.f = f;
  unsigned r = v.u + 0x7FFFu + ((v.u >> 16) & 1u);
  return (unsigned short)(r >> 16);
}

__device__ __forceinline__ float wave_reduce_add(float v) {
#pragma unroll
  for (int off = 16; off > 0; off >>= 1) v += __shfl_down(v, off, 32);
  return v;
}

// ---------------------------------------------------------------------------
// K0a: pack Ua (A x 2H fp32) -> bf16
// ---------------------------------------------------------------------------
__global__ __launch_bounds__(256) void pack_bf16_kernel(
    const float* __restrict__ src, unsigned short* __restrict__ dst, int n) {
  int i = blockIdx.x * 256 + threadIdx.x;
  if (i < n) dst[i] = f32_to_bf16(src[i]);
}

// ---------------------------------------------------------------------------
// K0b: dec[b,a] = sprev[b,:] . Wa[a,:]   (one wave per output)
// ---------------------------------------------------------------------------
__global__ __launch_bounds__(256) void dec_score_kernel(
    const float* __restrict__ sprev, const float* __restrict__ Wa,
    float* __restrict__ dec) {
  int gid  = blockIdx.x * 8 + (threadIdx.x >> 5);
  int lane = threadIdx.x & 31;
  int b = gid >> 9, a = gid & 511;
  const float* sb = sprev + b * HH;
  const float* wa = Wa + (size_t)a * HH;
  float sum = 0.f;
#pragma unroll
  for (int it = 0; it < 4; ++it) {
    int i = it * 128 + lane * 4;
    float4 s4 = *(const float4*)(sb + i);
    float4 w4 = *(const float4*)(wa + i);
    sum += s4.x * w4.x + s4.y * w4.y + s4.z * w4.z + s4.w * w4.w;
  }
  sum = wave_reduce_add(sum);
  if (lane == 0) dec[b * AA + a] = sum;
}

// ---------------------------------------------------------------------------
// K1: fused attention energies (137-GFLOP GEMM, bf16 WMMA, f32 accum)
//   energies[b,l] = sum_a va[a] * tanh(dec[b,a] + sum_h Ua[a,h]*enc[b,h,l])
//
// One workgroup per (b, 64-wide l block): 512 threads = 16 waves.
// Wave w owns M-tiles [2w, 2w+2) x 4 N-tiles -> 8 f32 accumulators.
//
// B tile is stored FRAGMENT-MAJOR in LDS: [kc][ntile][lane][16 halfs], so
// each wave's 8 ds_load_b128 per K-chunk are immediate offsets off one base
// register (bumped 4096 B/iter): no per-iteration VALU address math, no
// WMMA->VALU hazard NOPs, fully contiguous conflict-free 1KB fragment reads.
// ---------------------------------------------------------------------------
#define NBLK   64     // l-columns per workgroup (4 x 16-wide B tiles)

__global__ __launch_bounds__(512) void attn_energies_kernel(
    const float* __restrict__ enc,            // (B, 2H, L)
    const unsigned short* __restrict__ uabf,  // (A, 2H) bf16
    const float* __restrict__ dec,            // (B, A)
    const float* __restrict__ va,             // (A)
    float* __restrict__ energies) {           // (B, L)
  // [kc:32][n:4][lane:32][slot:16] ushorts = 128 KB
  __shared__ unsigned short sB[32 * 4 * 32 * 16];
  __shared__ float sDec[AA];
  __shared__ float sVa[AA];
  __shared__ float sE[NBLK];

  const int tid = threadIdx.x;
  const int b   = blockIdx.y;
  const int l0  = blockIdx.x * NBLK;

  // Stage enc tile (K=1024 x N=64) into LDS bf16, pre-swizzled into the
  // B-fragment order used here (k = 16*half + slot, as for A). Global reads stay fully coalesced
  // (consecutive threads -> consecutive l, 256B segments).
  const float* encb = enc + (size_t)b * H2 * LL + l0;
  for (int i = tid; i < H2 * NBLK; i += 512) {
    int k = i >> 6, col = i & 63;
    int kc = k >> 5, kk = k & 31;
    int ks = kk >> 4, slot = kk & 15;       // B frag: lane half, slot in run
    int n = col >> 4, c16 = col & 15;       // N tile, column within tile
    int fl = ks * 16 + c16;                 // fragment lane
    sB[(((kc * 4 + n) * 32) + fl) * 16 + slot] =
        f32_to_bf16(encb[(size_t)k * LL + col]);
  }
  for (int i = tid; i < AA; i += 512) {
    sDec[i] = dec[b * AA + i];
    sVa[i]  = va[i];
  }
  if (tid < NBLK) sE[tid] = 0.0f;
  __syncthreads();

  const int wave  = tid >> 5;
  const int lane  = tid & 31;
  const int row   = lane & 15;   // A-fragment: M row
  const int khalf = lane >> 4;   // A-fragment: which K-half
  const int col   = lane & 15;   // C: N column (within tile)
  const int ksel  = lane >> 4;   // C: row-half select

  union Frag { u32x4 u[2]; v16bf v; };

  v8f acc[2][4];
#pragma unroll
  for (int m = 0; m < 2; ++m)
#pragma unroll
    for (int n = 0; n < 4; ++n) acc[m][n] = (v8f){};

  const int mbase = wave * 32;  // wave covers A rows [mbase, mbase+32)
  const unsigned short* uaw  = uabf + (size_t)(mbase + row) * H2 + khalf * 16;
  const unsigned short* ldsb = &sB[lane * 16];  // + kc*2048 + n*512 (ushorts)

  for (int kc = 0; kc < 32; ++kc) {
    // prefetch A panel a few K-chunks ahead into WGP-level caches
    if (kc + 4 < 32) {
      __builtin_prefetch(uaw + (kc + 4) * 32, 0, 3);
      __builtin_prefetch(uaw + 16 * H2 + (kc + 4) * 32, 0, 3);
    }

    const unsigned short* pb = ldsb + kc * 2048;
    Frag bf[4];
#pragma unroll
    for (int n = 0; n < 4; ++n) {
      bf[n].u[0] = *(const u32x4*)(pb + n * 512);
      bf[n].u[1] = *(const u32x4*)(pb + n * 512 + 8);
    }

#pragma unroll
    for (int m = 0; m < 2; ++m) {
      Frag af;
      const unsigned short* pa = uaw + (size_t)m * 16 * H2 + kc * 32;
      af.u[0] = *(const u32x4*)(pa);
      af.u[1] = *(const u32x4*)(pa + 8);
      acc[m][0] = __builtin_amdgcn_wmma_f32_16x16x32_bf16(
          false, af.v, false, bf[0].v, (short)0, acc[m][0], false, false);
      acc[m][1] = __builtin_amdgcn_wmma_f32_16x16x32_bf16(
          false, af.v, false, bf[1].v, (short)0, acc[m][1], false, false);
      acc[m][2] = __builtin_amdgcn_wmma_f32_16x16x32_bf16(
          false, af.v, false, bf[2].v, (short)0, acc[m][2], false, false);
      acc[m][3] = __builtin_amdgcn_wmma_f32_16x16x32_bf16(
          false, af.v, false, bf[3].v, (short)0, acc[m][3], false, false);
    }
  }

  // Epilogue: v = tanh(dec + s); energy per column = sum_a va[a]*v
  // C layout: slot r, lanes 0-15 -> M=r, lanes 16-31 -> M=r+8; N = lane&15.
  float ep[4] = {0.f, 0.f, 0.f, 0.f};
#pragma unroll
  for (int m = 0; m < 2; ++m) {
#pragma unroll
    for (int r = 0; r < 8; ++r) {
      int ai = mbase + m * 16 + 8 * ksel + r;
      float d = sDec[ai], w = sVa[ai];
      ep[0] += w * tanhf(d + acc[m][0][r]);
      ep[1] += w * tanhf(d + acc[m][1][r]);
      ep[2] += w * tanhf(d + acc[m][2][r]);
      ep[3] += w * tanhf(d + acc[m][3][r]);
    }
  }
#pragma unroll
  for (int n = 0; n < 4; ++n) atomicAdd(&sE[n * 16 + col], ep[n]);
  __syncthreads();
  if (tid < NBLK) energies[(size_t)b * LL + l0 + tid] = sE[tid];
}

// ---------------------------------------------------------------------------
// K2: softmax over L per batch row
// ---------------------------------------------------------------------------
__global__ __launch_bounds__(256) void softmax_kernel(
    const float* __restrict__ e, float* __restrict__ alpha) {
  __shared__ float red[256];
  const int b = blockIdx.x, tid = threadIdx.x;
  const float* er = e + (size_t)b * LL;
  float v[8], m = -1e30f;
#pragma unroll
  for (int i = 0; i < 8; ++i) { v[i] = er[tid + i * 256]; m = fmaxf(m, v[i]); }
  red[tid] = m; __syncthreads();
  for (int s = 128; s > 0; s >>= 1) {
    if (tid < s) red[tid] = fmaxf(red[tid], red[tid + s]);
    __syncthreads();
  }
  m = red[0]; __syncthreads();
  float sum = 0.f;
#pragma unroll
  for (int i = 0; i < 8; ++i) { v[i] = __expf(v[i] - m); sum += v[i]; }
  red[tid] = sum; __syncthreads();
  for (int s = 128; s > 0; s >>= 1) {
    if (tid < s) red[tid] += red[tid + s];
    __syncthreads();
  }
  float inv = 1.0f / red[0];
#pragma unroll
  for (int i = 0; i < 8; ++i) alpha[(size_t)b * LL + tid + i * 256] = v[i] * inv;
}

// ---------------------------------------------------------------------------
// K3: context c[b,h] = sum_l alpha[b,l] * enc[b,h,l]   (bandwidth bound)
// grid (2H/8, B); one wave per h
// ---------------------------------------------------------------------------
__global__ __launch_bounds__(256) void context_kernel(
    const float* __restrict__ alpha, const float* __restrict__ enc,
    float* __restrict__ c) {
  __shared__ float sA[LL];
  const int b = blockIdx.y, tid = threadIdx.x;
  for (int i = tid; i < LL; i += 256) sA[i] = alpha[(size_t)b * LL + i];
  __syncthreads();
  const int wave = tid >> 5, lane = tid & 31;
  const int h = blockIdx.x * 8 + wave;
  const float* row = enc + (size_t)b * H2 * LL + (size_t)h * LL;
  float sum = 0.f;
#pragma unroll 4
  for (int it = 0; it < 16; ++it) {
    int idx = it * 128 + lane * 4;
    float4 e4 = *(const float4*)(row + idx);
    float4 a4 = *(const float4*)(&sA[idx]);
    sum += a4.x * e4.x + a4.y * e4.y + a4.z * e4.z + a4.w * e4.w;
  }
  sum = wave_reduce_add(sum);
  if (lane == 0) c[b * H2 + h] = sum;
}

// ---------------------------------------------------------------------------
// K4a: r and z gates (+ rs = r*sprev); one wave per (gate,b,h)
// ---------------------------------------------------------------------------
__device__ __forceinline__ float gate_dot(const float* __restrict__ xv, const float* __restrict__ wv,
                                          int n, int lane) {
  float sum = 0.f;
  for (int it = 0; it < n / 128; ++it) {
    int i = it * 128 + lane * 4;
    float4 a = *(const float4*)(xv + i);
    float4 w = *(const float4*)(wv + i);
    sum += a.x * w.x + a.y * w.y + a.z * w.z + a.w * w.w;
  }
  return sum;
}

__global__ __launch_bounds__(256) void gates_rz_kernel(
    const float* __restrict__ x, const float* __restrict__ sprev,
    const float* __restrict__ c,
    const float* __restrict__ Wr, const float* __restrict__ Ur,
    const float* __restrict__ Cr, const float* __restrict__ br,
    const float* __restrict__ Wz, const float* __restrict__ Uz,
    const float* __restrict__ Cz, const float* __restrict__ bz,
    float* __restrict__ r_out, float* __restrict__ rs_out,
    float* __restrict__ z_out) {
  const int gid  = blockIdx.x * 8 + (threadIdx.x >> 5);
  const int lane = threadIdx.x & 31;
  const int gate = gid >> 15;           // 0 = r, 1 = z
  const int rem  = gid & 32767;
  const int b = rem >> 9, h = rem & 511;

  float sum = gate_dot(x + (size_t)b * 512, (gate ? Wz : Wr) + (size_t)h * 512, 512, lane);
  sum += gate_dot(sprev + (size_t)b * HH, (gate ? Uz : Ur) + (size_t)h * HH, HH, lane);
  sum += gate_dot(c + (size_t)b * H2, (gate ? Cz : Cr) + (size_t)h * H2, H2, lane);
  sum = wave_reduce_add(sum);
  if (lane == 0) {
    sum += (gate ? bz : br)[h];
    float g = 1.0f / (1.0f + __expf(-sum));
    if (gate == 0) {
      r_out[b * HH + h]  = g;
      rs_out[b * HH + h] = g * sprev[b * HH + h];
    } else {
      z_out[b * HH + h] = g;
    }
  }
}

// ---------------------------------------------------------------------------
// K4b: s_prop + final blend; one wave per (b,h)
// ---------------------------------------------------------------------------
__global__ __launch_bounds__(256) void gates_out_kernel(
    const float* __restrict__ x, const float* __restrict__ sprev,
    const float* __restrict__ c, const float* __restrict__ rs,
    const float* __restrict__ z,
    const float* __restrict__ Ws, const float* __restrict__ Us,
    const float* __restrict__ Cs, const float* __restrict__ bs,
    float* __restrict__ out) {
  const int gid  = blockIdx.x * 8 + (threadIdx.x >> 5);
  const int lane = threadIdx.x & 31;
  const int b = gid >> 9, h = gid & 511;

  float sum = gate_dot(x + (size_t)b * 512, Ws + (size_t)h * 512, 512, lane);
  sum += gate_dot(rs + (size_t)b * HH, Us + (size_t)h * HH, HH, lane);
  sum += gate_dot(c + (size_t)b * H2, Cs + (size_t)h * H2, H2, lane);
  sum = wave_reduce_add(sum);
  if (lane == 0) {
    float sp = tanhf(sum + bs[h]);
    float zz = z[b * HH + h];
    out[b * HH + h] = zz * sp + (1.0f - zz) * sprev[b * HH + h];
  }
}

// ---------------------------------------------------------------------------
// Launch
// ---------------------------------------------------------------------------
extern "C" void kernel_launch(void* const* d_in, const int* in_sizes, int n_in,
                              void* d_out, int out_size, void* d_ws, size_t ws_size,
                              hipStream_t stream) {
  const float* x     = (const float*)d_in[0];
  const float* sprev = (const float*)d_in[1];
  const float* enc   = (const float*)d_in[2];
  const float* Ws    = (const float*)d_in[3];
  const float* Wz    = (const float*)d_in[4];
  const float* Wr    = (const float*)d_in[5];
  const float* Us    = (const float*)d_in[6];
  const float* Uz    = (const float*)d_in[7];
  const float* Ur    = (const float*)d_in[8];
  const float* Cs    = (const float*)d_in[9];
  const float* Cz    = (const float*)d_in[10];
  const float* Cr    = (const float*)d_in[11];
  const float* bs    = (const float*)d_in[12];
  const float* bz    = (const float*)d_in[13];
  const float* br    = (const float*)d_in[14];
  const float* va    = (const float*)d_in[15];
  const float* Wa    = (const float*)d_in[16];
  const float* Ua    = (const float*)d_in[17];
  float* out = (float*)d_out;

  // workspace layout (bytes)
  char* w = (char*)d_ws;
  unsigned short* ua_bf = (unsigned short*)(w);                 // 1,048,576
  float* dec      = (float*)(w + 1048576);                      //   131,072
  float* energies = (float*)(w + 1048576 + 131072);             //   524,288
  float* alpha    = (float*)(w + 1048576 + 131072 + 524288);    //   524,288
  float* ctx      = (float*)(w + 1048576 + 131072 + 2 * 524288);//   262,144
  float* r_buf    = (float*)(w + 1048576 + 131072 + 2 * 524288 + 262144);
  float* z_buf    = r_buf + BB * HH;
  float* rs_buf   = z_buf + BB * HH;

  // K0a: Ua -> bf16 (A*2H = 524288 elements)
  pack_bf16_kernel<<<(AA * H2 + 255) / 256, 256, 0, stream>>>(Ua, ua_bf, AA * H2);
  // K0b: dec scores (64*512 waves)
  dec_score_kernel<<<(BB * AA) / 8, 256, 0, stream>>>(sprev, Wa, dec);
  // K1: fused attention energies (WMMA bf16), 64-wide l blocks
  attn_energies_kernel<<<dim3(LL / NBLK, BB), 512, 0, stream>>>(enc, ua_bf, dec, va, energies);
  // K2: softmax
  softmax_kernel<<<BB, 256, 0, stream>>>(energies, alpha);
  // K3: context
  context_kernel<<<dim3(H2 / 8, BB), 256, 0, stream>>>(alpha, enc, ctx);
  // K4a: r, z gates (2*64*512 waves)
  gates_rz_kernel<<<(2 * BB * HH) / 8, 256, 0, stream>>>(
      x, sprev, ctx, Wr, Ur, Cr, br, Wz, Uz, Cz, bz, r_buf, rs_buf, z_buf);
  // K4b: s_prop + output blend
  gates_out_kernel<<<(BB * HH) / 8, 256, 0, stream>>>(
      x, sprev, ctx, rs_buf, z_buf, Ws, Us, Cs, bs, out);
}
